// GATv2Pytorch_70282844832158
// MI455X (gfx1250) — matrix operand order repaired and checked by reading, not compiled, not run on hardware
//
#include <hip/hip_runtime.h>

// GATv2 fused pipeline for MI455X (gfx1250, wave32, WMMA).
// B=2, N=512, FIN=128, H=4, D=64.

#define BB  2
#define NN  512
#define FINX 128
#define HH  4
#define DD  64
#define NEG_SLOPE 0.2f

typedef _Float16 f16;
typedef __attribute__((ext_vector_type(16))) _Float16 v16h;
typedef __attribute__((ext_vector_type(8)))  _Float16 v8h;
typedef __attribute__((ext_vector_type(8)))  float    v8f;

union V16U { v16h v; v8h h[2]; };

static __device__ __forceinline__ v8f wmma_f16(v16h a, v16h b, v8f c) {
  // D = A(16x32 f16) x B(32x16 f16) + C(16x16 f32)
  return __builtin_amdgcn_wmma_f32_16x16x32_f16(
      /*neg_a=*/false, a, /*neg_b=*/false, b,
      /*c_mod=*/(short)0, c, /*reuse_a=*/false, /*reuse_b=*/false);
}

// ---------------------------------------------------------------------------
// K0: precision conversion + layout prep (tiny, one pass).
//   xh   [B][N][FIN]  f16   (A-operand layout: row-major over K=FIN)
//   WpT  [H][D][FIN]  f16   (B-operand layout: contiguous K at fixed d)
//   Wc16 [H][D][2D]   f16   (native layout already B-friendly: contiguous d)
// ---------------------------------------------------------------------------
__global__ void k_prep(const float* __restrict__ x, const float* __restrict__ Wp,
                       const float* __restrict__ Wc,
                       f16* __restrict__ xh, f16* __restrict__ WpT,
                       f16* __restrict__ Wc16) {
  int t = blockIdx.x * blockDim.x + threadIdx.x;
  const int NX = BB * NN * FINX;       // 131072
  const int NW = HH * FINX * DD;       // 32768
  const int NC = HH * DD * 2 * DD;     // 32768
  if (t < NX) { xh[t] = (f16)x[t]; return; }
  t -= NX;
  if (t < NW) {  // WpT[h][d][i] = Wp[h][i][d]
    int h = t / (DD * FINX);
    int r = t % (DD * FINX);
    int d = r / FINX;
    int i = r % FINX;
    WpT[t] = (f16)Wp[(h * FINX + i) * DD + d];
    return;
  }
  t -= NW;
  if (t < NC) Wc16[t] = (f16)Wc[t];
}

// ---------------------------------------------------------------------------
// K1: h = x @ W_proj + b_proj   (per (b,h): 512x128 @ 128x64)
// One wave computes a 16x64 strip via 4 N-tiles x 4 K-chunks of WMMA.
// Writes h16 [bh][n][d] (A-operand for K2) and hT16 [bh][d][n]
// (B-operand for the attention AV matmul in K3).
// ---------------------------------------------------------------------------
__global__ void __launch_bounds__(32) k_proj(
    const f16* __restrict__ xh, const f16* __restrict__ WpT,
    const float* __restrict__ b_proj,
    f16* __restrict__ h16, f16* __restrict__ hT16) {
  const int bh = blockIdx.x;            // 0..7
  const int b  = bh >> 2, hh = bh & 3;
  const int i0 = blockIdx.y * 16;
  const int l  = threadIdx.x;           // 0..31 (wave32)
  const int lm = l & 15, lh = l >> 4;

  v8f acc[4] = {};
  const f16* arow = xh + (size_t)(b * NN + i0 + lm) * FINX;

#pragma unroll
  for (int kc = 0; kc < 4; ++kc) {
    const int k0 = kc * 32;
    V16U A;  // A 16x32: lane holds row lm, contiguous K run of 16 at 16*lh
    A.h[0] = *(const v8h*)(arow + k0 + 16 * lh);
    A.h[1] = *(const v8h*)(arow + k0 + 16 * lh + 8);
#pragma unroll
    for (int t = 0; t < 4; ++t) {
      // B 32x16: lane holds col (t*16+lm), contiguous K run of 16 at 16*lh
      const f16* brow = WpT + (size_t)(hh * DD + t * 16 + lm) * FINX + k0 + 16 * lh;
      v16h Bv = *(const v16h*)brow;
      acc[t] = wmma_f16(A.v, Bv, acc[t]);
    }
  }

#pragma unroll
  for (int t = 0; t < 4; ++t) {
    const int d = t * 16 + lm;
    const float bp = b_proj[hh * DD + d];
#pragma unroll
    for (int r = 0; r < 8; ++r) {
      const int n = i0 + r + 8 * lh;    // C layout: VGPR r -> row r + 8*(l/16)
      f16 hv = (f16)(acc[t][r] + bp);
      h16[((size_t)bh * NN + n) * DD + d]  = hv;
      hT16[((size_t)bh * DD + d) * NN + n] = hv;
    }
  }
}

// ---------------------------------------------------------------------------
// K2: zi = h @ W1^T (+W_cat_bias folded), zj = h @ W2^T   (512x64 @ 64x64 x2)
// W_cat_weight's native [h][e][d] layout is exactly the B-operand layout.
// ---------------------------------------------------------------------------
__global__ void __launch_bounds__(32) k_z(
    const f16* __restrict__ h16, const f16* __restrict__ Wc16,
    const float* __restrict__ Wc_bias,
    float* __restrict__ zib, float* __restrict__ zjw) {
  const int bh = blockIdx.x;
  const int hh = bh & 3;
  const int i0 = blockIdx.y * 16;
  const int l = threadIdx.x, lm = l & 15, lh = l >> 4;

  v8f ai[4] = {}, aj[4] = {};
  const f16* arow = h16 + ((size_t)bh * NN + i0 + lm) * DD;

#pragma unroll
  for (int kc = 0; kc < 2; ++kc) {
    const int k0 = kc * 32;
    V16U A;
    A.h[0] = *(const v8h*)(arow + k0 + 16 * lh);
    A.h[1] = *(const v8h*)(arow + k0 + 16 * lh + 8);
#pragma unroll
    for (int t = 0; t < 4; ++t) {
      const f16* brow = Wc16 + (size_t)(hh * DD + t * 16 + lm) * (2 * DD);
      v16h B1 = *(const v16h*)(brow + k0 + 16 * lh);        // W1 = Wc[:, :, :D]
      v16h B2 = *(const v16h*)(brow + DD + k0 + 16 * lh);   // W2 = Wc[:, :, D:]
      ai[t] = wmma_f16(A.v, B1, ai[t]);
      aj[t] = wmma_f16(A.v, B2, aj[t]);
    }
  }

#pragma unroll
  for (int t = 0; t < 4; ++t) {
    const int e = t * 16 + lm;
    const float wb = Wc_bias[hh * DD + e];
#pragma unroll
    for (int r = 0; r < 8; ++r) {
      const int n = i0 + r + 8 * lh;
      const size_t off = ((size_t)bh * NN + n) * DD + e;
      zib[off] = ai[t][r] + wb;   // bias folded into zi
      zjw[off] = aj[t][r];
    }
  }
}

// ---------------------------------------------------------------------------
// K2b: rank-1 terms of the LeakyReLU decomposition.
//   LReLU(s) = 0.6 s + 0.4 |s|  (slope 0.2)
//   u[i] = a . zib[i],  v[j] = a . zj[j]
// ---------------------------------------------------------------------------
__global__ void k_uv(const float* __restrict__ zib, const float* __restrict__ zjw,
                     const float* __restrict__ a,
                     float* __restrict__ u, float* __restrict__ v) {
  int t = blockIdx.x * blockDim.x + threadIdx.x;   // 0..B*H*N-1
  if (t >= BB * HH * NN) return;
  const int hh = (t / NN) & 3;
  const float* ap = a + hh * DD;
  const float* zi = zib + (size_t)t * DD;
  const float* zj = zjw + (size_t)t * DD;
  float su = 0.f, sv = 0.f;
#pragma unroll
  for (int e = 0; e < DD; ++e) {
    su = fmaf(ap[e], zi[e], su);
    sv = fmaf(ap[e], zj[e], sv);
  }
  u[t] = su;
  v[t] = sv;
}

// ---------------------------------------------------------------------------
// K3: fused flash-style attention. One wave per (b,h, 16-row i-tile).
// Streams j in chunks of 32 (one j per lane):
//   e[i,j] = 0.6*(u_i + v_j) + 0.4 * sum_e a_e * |zib[i,e] + zj[j,e]|
//   online softmax (wave shuffles), P staged in LDS as WMMA A-operand,
//   O += P(16x32 f16) @ H(32x64 f16) via 4 x v_wmma_f32_16x16x32_f16.
// Never materializes the 8MB e matrix.
// ---------------------------------------------------------------------------
__global__ void __launch_bounds__(32) k_attn(
    const float* __restrict__ zib, const float* __restrict__ zjw,
    const float* __restrict__ u, const float* __restrict__ v,
    const float* __restrict__ a, const f16* __restrict__ hT16,
    const float* __restrict__ bias_param, float* __restrict__ out) {
  const int bh = blockIdx.x;
  const int b = bh >> 2, hh = bh & 3;
  const int i0 = blockIdx.y * 16;
  const int l = threadIdx.x, lm = l & 15, lh = l >> 4;

  __shared__ float zib_s[16 * DD];
  __shared__ float u_s[16], m_s[16], l_s[16], sc_s[16];
  __shared__ __align__(32) f16 p_s[16 * 32];

  // a[h,:] into registers (uniform, 64 VGPRs — plenty at 1 wave/SIMD occupancy)
  float a_r[DD];
  {
    const float* ap = a + hh * DD;
#pragma unroll
    for (int e = 0; e < DD; ++e) a_r[e] = ap[e];
  }
  {  // zib tile (16x64 f32) into LDS
    const float* src = zib + ((size_t)bh * NN + i0) * DD;
    for (int k = l; k < 16 * DD; k += 32) zib_s[k] = src[k];
  }
  if (l < 16) {
    u_s[l] = u[(size_t)bh * NN + i0 + l];
    m_s[l] = -3.0e38f;
    l_s[l] = 0.f;
  }
  __syncthreads();

  v8f acc[4] = {};   // O tile: 16 rows x 64 cols f32

  for (int c = 0; c < NN / 32; ++c) {
    const int j0 = c * 32;
    const int j = j0 + l;

    float zj_r[DD];   // this lane's zj row
    {
      const float4* zp = (const float4*)(zjw + ((size_t)bh * NN + j) * DD);
#pragma unroll
      for (int q = 0; q < DD / 4; ++q) {
        float4 t4 = zp[q];
        zj_r[4 * q + 0] = t4.x; zj_r[4 * q + 1] = t4.y;
        zj_r[4 * q + 2] = t4.z; zj_r[4 * q + 3] = t4.w;
      }
    }
    const float v_l = v[(size_t)bh * NN + j];

    for (int i = 0; i < 16; ++i) {
      const float* zr = &zib_s[i * DD];
      float w = 0.f;
#pragma unroll
      for (int e = 0; e < DD; ++e) {
        float s = zr[e] + zj_r[e];          // ds broadcast + v_add
        w = fmaf(a_r[e], fabsf(s), w);      // v_fma with |src| modifier
      }
      float esc = 0.6f * (u_s[i] + v_l) + 0.4f * w;

      float mc = esc;                        // wave32 max-reduce
#pragma unroll
      for (int mm = 16; mm > 0; mm >>= 1) mc = fmaxf(mc, __shfl_xor(mc, mm, 32));
      const float mold = m_s[i];
      const float mnew = fmaxf(mold, mc);
      const float p = __expf(esc - mnew);
      float ps = p;                          // wave32 sum-reduce
#pragma unroll
      for (int mm = 16; mm > 0; mm >>= 1) ps += __shfl_xor(ps, mm, 32);
      const float scale = __expf(mold - mnew);
      if (l == 0) { m_s[i] = mnew; sc_s[i] = scale; l_s[i] = l_s[i] * scale + ps; }
      p_s[i * 32 + l] = (f16)p;
    }
    __syncthreads();

    // rescale running O by exp(m_old - m_new), per row
#pragma unroll
    for (int t = 0; t < 4; ++t)
#pragma unroll
      for (int r = 0; r < 8; ++r) acc[t][r] *= sc_s[r + 8 * lh];

    // A = P tile from LDS (16x32 f16, A-operand layout)
    V16U A;
    A.h[0] = *(const v8h*)&p_s[lm * 32 + 16 * lh];
    A.h[1] = *(const v8h*)&p_s[lm * 32 + 16 * lh + 8];

    // B tiles = h^T chunk (K=j contiguous); O += P @ H
#pragma unroll
    for (int t = 0; t < 4; ++t) {
      const f16* brow = hT16 + ((size_t)bh * DD + t * 16 + lm) * NN + j0 + 16 * lh;
      v16h Bv = *(const v16h*)brow;
      acc[t] = wmma_f16(A.v, Bv, acc[t]);
    }
    __syncthreads();
  }

  // epilogue: out[b, i, h*D + d] = O[i,d]/l_i + bias_param
#pragma unroll
  for (int t = 0; t < 4; ++t) {
    const int d = t * 16 + lm;
    const float bp = bias_param[hh * DD + d];
#pragma unroll
    for (int r = 0; r < 8; ++r) {
      const int row = r + 8 * lh;
      const int i = i0 + row;
      const float val = acc[t][r] / l_s[row] + bp;
      out[((size_t)(b * NN + i)) * (HH * DD) + hh * DD + d] = val;
    }
  }
}

// ---------------------------------------------------------------------------
extern "C" void kernel_launch(void* const* d_in, const int* in_sizes, int n_in,
                              void* d_out, int out_size, void* d_ws, size_t ws_size,
                              hipStream_t stream) {
  (void)in_sizes; (void)n_in; (void)out_size; (void)ws_size;
  const float* x      = (const float*)d_in[0];  // (B,N,FIN)
  const float* Wp     = (const float*)d_in[1];  // (H,FIN,D)
  const float* bp     = (const float*)d_in[2];  // (H,D)
  const float* Wc     = (const float*)d_in[3];  // (H,D,2D)
  const float* Wcb    = (const float*)d_in[4];  // (H,D)
  const float* a      = (const float*)d_in[5];  // (H,D)
  const float* bparam = (const float*)d_in[6];  // (H*D,)
  float* out = (float*)d_out;

  char* ws = (char*)d_ws;
  size_t off = 0;
  auto alloc = [&](size_t bytes) -> void* {
    void* p = ws + off;
    off += (bytes + 255) & ~(size_t)255;
    return p;
  };
  f16*   xh   = (f16*)alloc((size_t)BB * NN * FINX * sizeof(f16));
  f16*   WpT  = (f16*)alloc((size_t)HH * DD * FINX * sizeof(f16));
  f16*   Wc16 = (f16*)alloc((size_t)HH * DD * 2 * DD * sizeof(f16));
  f16*   h16  = (f16*)alloc((size_t)BB * HH * NN * DD * sizeof(f16));
  f16*   hT16 = (f16*)alloc((size_t)BB * HH * NN * DD * sizeof(f16));
  float* zib  = (float*)alloc((size_t)BB * HH * NN * DD * sizeof(float));
  float* zjw  = (float*)alloc((size_t)BB * HH * NN * DD * sizeof(float));
  float* u    = (float*)alloc((size_t)BB * HH * NN * sizeof(float));
  float* v    = (float*)alloc((size_t)BB * HH * NN * sizeof(float));

  const int prep_elems = BB * NN * FINX + HH * FINX * DD + HH * DD * 2 * DD;
  k_prep<<<dim3((prep_elems + 255) / 256), 256, 0, stream>>>(x, Wp, Wc, xh, WpT, Wc16);
  k_proj<<<dim3(BB * HH, NN / 16), 32, 0, stream>>>(xh, WpT, bp, h16, hT16);
  k_z<<<dim3(BB * HH, NN / 16), 32, 0, stream>>>(h16, Wc16, Wcb, zib, zjw);
  k_uv<<<dim3((BB * HH * NN + 255) / 256), 256, 0, stream>>>(zib, zjw, a, u, v);
  k_attn<<<dim3(BB * HH, NN / 16), 32, 0, stream>>>(zib, zjw, u, v, a, hT16, bparam, out);
}
